// PerceptualLoss_59055800320366
// MI455X (gfx1250) — matrix operand order repaired and checked by reading, not compiled, not run on hardware
//
#include <hip/hip_runtime.h>
#include <cstdint>

#define CCH 512

typedef __attribute__((ext_vector_type(16))) __bf16          v16bf;
typedef __attribute__((ext_vector_type(16))) unsigned short  v16u;
typedef __attribute__((ext_vector_type(8)))  float           v8f;
typedef __attribute__((ext_vector_type(2)))  float           v2f;
typedef __attribute__((ext_vector_type(4)))  unsigned int    u32x4;
typedef __attribute__((ext_vector_type(8)))  int             i32x8;
typedef __attribute__((ext_vector_type(4)))  int             i32x4;

#if defined(__has_builtin)
#  if __has_builtin(__builtin_amdgcn_tensor_load_to_lds)
#    define HAVE_TDM 1
#  endif
#endif
#ifndef HAVE_TDM
#  define HAVE_TDM 0
#endif

static __device__ __forceinline__ unsigned short f2bfbits(float f) {
  unsigned int u = __float_as_uint(f);
  unsigned int r = (u + 0x7FFFu + ((u >> 16) & 1u)) >> 16;
  return (unsigned short)r;
}

#if HAVE_TDM
// 2D tile load via Tensor Data Mover: tile = 64 rows x 32 bf16 (contiguous),
// row stride Ktot elems, data_size=2B. Descriptor per CDNA5 ISA ch.8.
static __device__ __forceinline__ void tdm_load_tile(const unsigned short* src,
                                                     int Ktot, unsigned ldsOff) {
  unsigned long long ga = (unsigned long long)(uintptr_t)src;
  u32x4 g0;
  g0[0] = 1u;                                           // count=1, user mode
  g0[1] = ldsOff;                                       // lds_addr
  g0[2] = (unsigned)(ga & 0xFFFFFFFFull);               // global_addr[31:0]
  g0[3] = (unsigned)((ga >> 32) & 0x01FFFFFFull)        // global_addr[56:32]
          | 0x80000000u;                                // type=2 (image)
  i32x8 g1;
  g1[0] = 0x00010000;                                   // data_size=1 (2 bytes)
  g1[1] = (int)(((unsigned)Ktot & 0xFFFFu) << 16);      // tensor_dim0 lo16
  g1[2] = (int)((512u << 16) | (((unsigned)Ktot >> 16) & 0xFFFFu)); // dim0 hi | dim1 lo
  g1[3] = (int)(32u << 16);                             // dim1 hi=0 | tile_dim0=32
  g1[4] = 64;                                           // tile_dim1=64, tile_dim2=0
  g1[5] = Ktot;                                         // tensor_dim0_stride lo32
  g1[6] = 0;                                            // stride hi16 | dim1_stride lo16
  g1[7] = 0;
  i32x4 z4 = {0, 0, 0, 0};
#if __clang_major__ >= 23
  i32x8 z8 = {0, 0, 0, 0, 0, 0, 0, 0};
  __builtin_amdgcn_tensor_load_to_lds(g0, g1, z4, z4, z8, 0);
#else
  __builtin_amdgcn_tensor_load_to_lds(g0, g1, z4, z4, 0);
#endif
}
#endif

// ---------------- Layer 0: cin=1, k=10, s=5 (tiny reduction, plain VALU) ----
__global__ __launch_bounds__(256)
void conv0_kernel(const float* __restrict__ wav, const float* __restrict__ W0,
                  float* __restrict__ out, int B, int Tin, int Tout) {
  long i = (long)blockIdx.x * blockDim.x + threadIdx.x;
  long total = (long)B * CCH * Tout;
  if (i >= total) return;
  int t = (int)(i % Tout);
  long tmp = i / Tout;
  int co = (int)(tmp % CCH);
  int b  = (int)(tmp / CCH);
  const float* w = W0 + co * 10;
  const float* x = wav + (long)b * Tin + (long)t * 5;
  float acc = 0.f;
#pragma unroll
  for (int j = 0; j < 10; ++j) acc += w[j] * x[j];
  out[i] = acc;
}

// ---------------- GroupNorm(1 group) statistics, one block per batch --------
__global__ __launch_bounds__(256)
void gn_stats_kernel(const float* __restrict__ x, long CT,
                     float* __restrict__ mu, float* __restrict__ rstd) {
  int b = blockIdx.x;
  const float* xb = x + (long)b * CT;
  double s = 0.0, ss = 0.0;
  for (long i = threadIdx.x; i < CT; i += blockDim.x) {
    float v = xb[i];
    s += v; ss += (double)v * v;
  }
  __shared__ double sh[256], sh2[256];
  sh[threadIdx.x] = s; sh2[threadIdx.x] = ss;
  __syncthreads();
  for (int off = 128; off > 0; off >>= 1) {
    if ((int)threadIdx.x < off) {
      sh[threadIdx.x]  += sh[threadIdx.x + off];
      sh2[threadIdx.x] += sh2[threadIdx.x + off];
    }
    __syncthreads();
  }
  if (threadIdx.x == 0) {
    double m = sh[0] / (double)CT;
    double var = sh2[0] / (double)CT - m * m;
    mu[b]   = (float)m;
    rstd[b] = rsqrtf((float)var + 1e-5f);
  }
}

// ---------------- GN apply + ReLU; write bf16 feed and/or f32 features ------
__global__ __launch_bounds__(256)
void gn_apply_kernel(const float* __restrict__ x, const float* __restrict__ mu,
                     const float* __restrict__ rstd, const float* __restrict__ gamma,
                     const float* __restrict__ beta, unsigned short* __restrict__ obf,
                     float* __restrict__ of32, int Tout, long total) {
  long i = (long)blockIdx.x * blockDim.x + threadIdx.x;
  if (i >= total) return;
  int c = (int)((i / Tout) % CCH);
  int b = (int)(i / ((long)Tout * CCH));
  float v = (x[i] - mu[b]) * rstd[b] * gamma[c] + beta[c];
  v = fmaxf(v, 0.f);
  if (obf)  obf[i]  = f2bfbits(v);
  if (of32) of32[i] = v;
}

// ---------------- f32 -> bf16 cast (weight packing; layout already [co][K]) -
__global__ __launch_bounds__(256)
void cast_bf16_kernel(const float* __restrict__ src, unsigned short* __restrict__ dst, long n) {
  long i = (long)blockIdx.x * blockDim.x + threadIdx.x;
  if (i < n) dst[i] = f2bfbits(src[i]);
}

// ---------------- Implicit-GEMM conv via v_wmma_f32_16x16x32_bf16 -----------
// D[co, p] = sum_{K=ci*k+j} W[co,K] * h[b(p), ci, t(p)*s + j]
// Block: 256 thr = 8 waves. Block tile 64 co x 64 pos; wave tile 16 co x 32 pos
// (2 accumulators -> A-fragment reuse). A tile double-buffered via TDM.
__global__ __launch_bounds__(256)
void conv_wmma_kernel(const unsigned short* __restrict__ h,   // [B][512][Tin] bf16 bits
                      const unsigned short* __restrict__ Wp,  // [512][Ktot]   bf16 bits
                      float* __restrict__ out,                // [B][512][Tout]
                      int B, int Tin, int Tout, int kshift, int sshift) {
  const int Ktot = CCH << kshift;
  const int km1  = (1 << kshift) - 1;
  const int coBlk = blockIdx.x * 64;
  const long pBlk = (long)blockIdx.y * 64;
  const int tid  = threadIdx.x;
  const int wave = tid >> 5;
  const int lane = tid & 31;
  const int coSub = (wave >> 1) * 16;    // 0,16,32,48
  const int pSub  = (wave & 1) * 32;     // 0,32
  const int hi  = lane >> 4;
  const int l15 = lane & 15;
  const long NP = (long)B * Tout;

  __shared__ unsigned short As[2][64][32];   // [buf][co][K], TDM destination
  __shared__ unsigned short Bs[32][66];      // [K][pos], padded

  v8f acc[2] = {{}, {}};

#if HAVE_TDM
  const bool tdmWave = (wave == 0);
  if (tdmWave)
    tdm_load_tile(Wp + (long)coBlk * Ktot, Ktot,
                  (unsigned)(uintptr_t)(void*)&As[0][0][0]);
#endif

  int buf = 0;
  for (int kk = 0; kk < Ktot; kk += 32) {
#if !HAVE_TDM
    // Manual A staging: 64 co x 32 K (2048 elems, 8 contiguous per thread)
    {
      int r = tid >> 2;
      int c = (tid & 3) * 8;
      const unsigned short* src = Wp + (long)(coBlk + r) * Ktot + kk + c;
#pragma unroll
      for (int e = 0; e < 8; ++e) As[0][r][c + e] = src[e];
    }
#endif
    // Stage B tile (im2col): 32 K x 64 pos (2048 elems, 8 per thread)
    {
      int row  = (tid * 8) >> 6;
      int col0 = (tid * 8) & 63;
      int K  = kk + row;
      int ci = K >> kshift;
      int j  = K & km1;
      const unsigned short* hb = h + (long)ci * Tin;
#pragma unroll
      for (int e = 0; e < 8; ++e) {
        long p = pBlk + col0 + e;
        unsigned short v = 0;
        if (p < NP) {
          int b = (int)(p / Tout);
          int t = (int)(p - (long)b * Tout);
          v = hb[(long)b * CCH * Tin + ((long)t << sshift) + j];
        }
        Bs[row][col0 + e] = v;
      }
      // Prefetch next K-chunk's row for this thread (global_prefetch_b8)
      if (kk + 32 < Ktot) {
        long p = pBlk + col0;
        if (p < NP) {
          int b = (int)(p / Tout);
          int t = (int)(p - (long)b * Tout);
          int Kn  = kk + 32 + row;
          int cin = Kn >> kshift;
          int jn  = Kn & km1;
          __builtin_prefetch(h + (long)cin * Tin + (long)b * CCH * Tin +
                             ((long)t << sshift) + jn, 0, 3);
        }
      }
    }
#if HAVE_TDM
    // Issue next A tile into the alternate buffer, drain the current one.
    if (tdmWave) {
      if (kk + 32 < Ktot) {
        tdm_load_tile(Wp + (long)coBlk * Ktot + (kk + 32), Ktot,
                      (unsigned)(uintptr_t)(void*)&As[buf ^ 1][0][0]);
        __builtin_amdgcn_s_wait_tensorcnt(1);
      } else {
        __builtin_amdgcn_s_wait_tensorcnt(0);
      }
    }
#endif
    __syncthreads();

#if HAVE_TDM
    const unsigned short (*Ac)[32] = As[buf];
#else
    const unsigned short (*Ac)[32] = As[0];
#endif
    // A fragment (16-bit A 16x32 layout): row = co, K split per lane half
    v16u au;
    {
      int row = coSub + l15;
#pragma unroll
      for (int e = 0; e < 16; ++e) {
        int K = ((e < 8) ? e : e + 8) + hi * 8;
        au[e] = Ac[row][K];
      }
    }
    v16bf a = __builtin_bit_cast(v16bf, au);
#pragma unroll
    for (int tt = 0; tt < 2; ++tt) {
      v16u bu;
      int col = pSub + tt * 16 + l15;
#pragma unroll
      for (int e = 0; e < 16; ++e) {
        bu[e] = Bs[((e < 8) ? e : e + 8) + hi * 8][col];
      }
      v16bf b = __builtin_bit_cast(v16bf, bu);
      acc[tt] = __builtin_amdgcn_wmma_f32_16x16x32_bf16(false, a, false, b,
                                                        (short)0, acc[tt], false, false);
    }
    __syncthreads();
#if HAVE_TDM
    buf ^= 1;
#endif
  }

  // D layout: lane -> N = l15, VGPR r -> M = r + hi*8
#pragma unroll
  for (int tt = 0; tt < 2; ++tt) {
    long p = pBlk + pSub + tt * 16 + l15;
    if (p < NP) {
      int b = (int)(p / Tout);
      int t = (int)(p - (long)b * Tout);
#pragma unroll
      for (int r = 0; r < 8; ++r) {
        int co = coBlk + coSub + r + hi * 8;
        out[((long)b * CCH + co) * Tout + t] = acc[tt][r];
      }
    }
  }
}

// ---------------- per-point squared norms -----------------------------------
__global__ __launch_bounds__(256)
void rownorm_kernel(const float* __restrict__ f, float* __restrict__ nn,
                    int D, int nrows) {
  int i = blockIdx.x * blockDim.x + threadIdx.x;
  if (i >= nrows) return;
  const float* r = f + (long)i * D;
  float s = 0.f;
  for (int d = 0; d < D; ++d) s += r[d] * r[d];
  nn[i] = s;
}

// ---------------- sq_cost via full-precision v_wmma_f32_16x16x4_f32 ---------
// C[b][n][m] = 0.5*max(|x_n|^2 + |y_m|^2 - 2 <x_n, y_m>, 0)
__global__ __launch_bounds__(128)
void sqcost_wmma_kernel(const float* __restrict__ X, const float* __restrict__ Y,
                        const float* __restrict__ nx, const float* __restrict__ ny,
                        float* __restrict__ Cm, int D, int B) {
  const int wave = threadIdx.x >> 5;
  const int lane = threadIdx.x & 31;
  const long tilesPerB = (CCH / 16) * (CCH / 16);
  long tile = (long)blockIdx.x * 4 + wave;
  if (tile >= (long)B * tilesPerB) return;
  int b  = (int)(tile / tilesPerB);
  int tr = (int)(tile % tilesPerB);
  int n0 = (tr / (CCH / 16)) * 16;
  int m0 = (tr % (CCH / 16)) * 16;
  const float* Xr = X + ((long)b * CCH + n0) * D;
  const float* Yr = Y + ((long)b * CCH + m0) * D;
  const int hi = lane >> 4, l15 = lane & 15;

  v8f acc = {};
  int Dal = (D + 3) & ~3;
  for (int kk = 0; kk < Dal; kk += 4) {
    int ka = kk + hi * 2;
    v2f a, bb;
    a[0]  = (ka     < D) ? Xr[(long)l15 * D + ka    ] : 0.f;
    a[1]  = (ka + 1 < D) ? Xr[(long)l15 * D + ka + 1] : 0.f;
    bb[0] = (ka     < D) ? Yr[(long)l15 * D + ka    ] : 0.f;
    bb[1] = (ka + 1 < D) ? Yr[(long)l15 * D + ka + 1] : 0.f;
    acc = __builtin_amdgcn_wmma_f32_16x16x4_f32(false, a, false, bb,
                                                (short)0, acc, false, false);
  }

  int m = m0 + l15;
  float bbn = ny[b * CCH + m];
  float* Cb = Cm + ((long)b * CCH) * CCH;
#pragma unroll
  for (int r = 0; r < 8; ++r) {
    int n = n0 + r + hi * 8;
    float c = nx[b * CCH + n] + bbn - 2.f * acc[r];
    Cb[(long)n * CCH + m] = 0.5f * fmaxf(c, 0.f);
  }
}

// ---------------- entropic OT (Sinkhorn in log domain), one block per batch -
__global__ __launch_bounds__(512)
void sinkhorn_kernel(const float* __restrict__ Cm, float* __restrict__ Sout,
                     float eps, int iters) {
  const int Np = CCH;
  int b = blockIdx.x;
  const float* Cb = Cm + (long)b * Np * Np;
  __shared__ float fsh[CCH], gsh[CCH], red[CCH];
  int tid = threadIdx.x;
  fsh[tid] = 0.f; gsh[tid] = 0.f;
  __syncthreads();
  const float lognp = -__logf((float)Np);
  const float inve  = 1.0f / eps;
  for (int it = 0; it < iters; ++it) {
    // g_m = -eps * LSE_n( loga + (f_n - C[n][m]) / eps )   (coalesced columns)
    {
      float mx = -3.4e38f, sum = 0.f;
      for (int n = 0; n < Np; ++n) {
        float t = lognp + (fsh[n] - Cb[(long)n * Np + tid]) * inve;
        if (t > mx) { sum = sum * __expf(mx - t) + 1.f; mx = t; }
        else        { sum += __expf(t - mx); }
      }
      float gnew = -eps * (mx + __logf(sum));
      __syncthreads();
      gsh[tid] = gnew;
      __syncthreads();
    }
    // f_n = -eps * LSE_m( logb + (g_m - C[n][m]) / eps )   (row per thread)
    {
      const float* Crow = Cb + (long)tid * Np;
      float mx = -3.4e38f, sum = 0.f;
      for (int m = 0; m < Np; ++m) {
        float t = lognp + (gsh[m] - Crow[m]) * inve;
        if (t > mx) { sum = sum * __expf(mx - t) + 1.f; mx = t; }
        else        { sum += __expf(t - mx); }
      }
      float fnew = -eps * (mx + __logf(sum));
      __syncthreads();
      fsh[tid] = fnew;
      __syncthreads();
    }
  }
  red[tid] = fsh[tid] + gsh[tid];
  __syncthreads();
  for (int off = 256; off > 0; off >>= 1) {
    if (tid < off) red[tid] += red[tid + off];
    __syncthreads();
  }
  if (tid == 0) Sout[b] = red[0] / (float)Np;
}

__global__ void combine_kernel(const float* __restrict__ Sxy, const float* __restrict__ Sxx,
                               const float* __restrict__ Syy, float* __restrict__ out, int B) {
  int b = blockIdx.x * blockDim.x + threadIdx.x;
  if (b < B) out[b] = Sxy[b] - 0.5f * (Sxx[b] + Syy[b]);
}

// ============================================================================
extern "C" void kernel_launch(void* const* d_in, const int* in_sizes, int n_in,
                              void* d_out, int out_size, void* d_ws, size_t ws_size,
                              hipStream_t stream) {
  (void)n_in; (void)out_size; (void)ws_size;
  const float* y_hat = (const float*)d_in[0];
  const float* ysig  = (const float*)d_in[1];
  const float* Wf[5]; const float* gf[5]; const float* bf[5];
  for (int i = 0; i < 5; ++i) {
    Wf[i] = (const float*)d_in[2 + 3 * i];
    gf[i] = (const float*)d_in[3 + 3 * i];
    bf[i] = (const float*)d_in[4 + 3 * i];
  }
  const int B = in_sizes[0] / 32000;
  const int T0 = 6399, T1 = 1598, T2 = 798, T3 = 398, T4 = 198;

  // ---- workspace carve-up
  size_t off = 0;
  char* ws = (char*)d_ws;
  auto take = [&](size_t bytes) { size_t o = off; off += (bytes + 255) & ~(size_t)255; return ws + o; };
  float*          bufF = (float*)         take((size_t)B * CCH * T0 * 4);   // conv out f32
  unsigned short* bufH = (unsigned short*)take((size_t)B * CCH * T0 * 2);   // normalized bf16
  float*          fx   = (float*)         take((size_t)B * CCH * T4 * 4);
  float*          fy   = (float*)         take((size_t)B * CCH * T4 * 4);
  unsigned short* Wp[4];
  const long wk[4] = {(long)CCH * CCH * 8, (long)CCH * CCH * 4,
                      (long)CCH * CCH * 4, (long)CCH * CCH * 4};
  for (int i = 0; i < 4; ++i) Wp[i] = (unsigned short*)take((size_t)wk[i] * 2);
  float* Cmat = (float*)take((size_t)B * CCH * CCH * 4);
  float* nx   = (float*)take((size_t)B * CCH * 4);
  float* ny   = (float*)take((size_t)B * CCH * 4);
  float* mu   = (float*)take((size_t)B * 4);
  float* rstd = (float*)take((size_t)B * 4);
  float* Sxy  = (float*)take((size_t)B * 4);
  float* Sxx  = (float*)take((size_t)B * 4);
  float* Syy  = (float*)take((size_t)B * 4);

  // ---- pack conv weights to bf16 (layout [co][ci*k+j] is already contiguous)
  for (int i = 0; i < 4; ++i)
    cast_bf16_kernel<<<(int)((wk[i] + 255) / 256), 256, 0, stream>>>(Wf[i + 1], Wp[i], wk[i]);

  // ---- feature extractor
  const int kshift[4] = {3, 2, 2, 2};   // k = 8,4,4,4
  const int sshift[4] = {2, 1, 1, 1};   // s = 4,2,2,2
  const int Tin[4] = {T0, T1, T2, T3};
  const int To[4]  = {T1, T2, T3, T4};

  auto run_features = [&](const float* wav, float* fout) {
    long tot0 = (long)B * CCH * T0;
    conv0_kernel<<<(int)((tot0 + 255) / 256), 256, 0, stream>>>(wav, Wf[0], bufF, B, 32000, T0);
    gn_stats_kernel<<<B, 256, 0, stream>>>(bufF, (long)CCH * T0, mu, rstd);
    gn_apply_kernel<<<(int)((tot0 + 255) / 256), 256, 0, stream>>>(
        bufF, mu, rstd, gf[0], bf[0], bufH, nullptr, T0, tot0);
    for (int i = 0; i < 4; ++i) {
      dim3 grid(CCH / 64, (unsigned)(((long)B * To[i] + 63) / 64));
      conv_wmma_kernel<<<grid, 256, 0, stream>>>(bufH, Wp[i], bufF, B, Tin[i], To[i],
                                                 kshift[i], sshift[i]);
      long tot = (long)B * CCH * To[i];
      gn_stats_kernel<<<B, 256, 0, stream>>>(bufF, (long)CCH * To[i], mu, rstd);
      gn_apply_kernel<<<(int)((tot + 255) / 256), 256, 0, stream>>>(
          bufF, mu, rstd, gf[i + 1], bf[i + 1],
          (i < 3) ? bufH : nullptr, (i == 3) ? fout : nullptr, To[i], tot);
    }
  };

  run_features(y_hat, fx);
  run_features(ysig, fy);

  // ---- Sinkhorn divergence
  const float eps = 0.05f * 0.05f;
  int nrows = B * CCH;
  rownorm_kernel<<<(nrows + 255) / 256, 256, 0, stream>>>(fx, nx, T4, nrows);
  rownorm_kernel<<<(nrows + 255) / 256, 256, 0, stream>>>(fy, ny, T4, nrows);
  long tiles = (long)B * (CCH / 16) * (CCH / 16);
  int sqBlocks = (int)((tiles + 3) / 4);

  sqcost_wmma_kernel<<<sqBlocks, 128, 0, stream>>>(fx, fy, nx, ny, Cmat, T4, B);
  sinkhorn_kernel<<<B, 512, 0, stream>>>(Cmat, Sxy, eps, 50);
  sqcost_wmma_kernel<<<sqBlocks, 128, 0, stream>>>(fx, fx, nx, nx, Cmat, T4, B);
  sinkhorn_kernel<<<B, 512, 0, stream>>>(Cmat, Sxx, eps, 50);
  sqcost_wmma_kernel<<<sqBlocks, 128, 0, stream>>>(fy, fy, ny, ny, Cmat, T4, B);
  sinkhorn_kernel<<<B, 512, 0, stream>>>(Cmat, Syy, eps, 50);

  combine_kernel<<<1, 256, 0, stream>>>(Sxy, Sxx, Syy, (float*)d_out, B);
}
